// ProteinBigBirdAttention_61280593379455
// MI455X (gfx1250) — matrix operand order repaired and checked by reading, not compiled, not run on hardware
//
#include <hip/hip_runtime.h>
#include <cstddef>
#include <cstdint>

#define BATCH   4
#define SEQ     4096
#define TOK     (BATCH*SEQ)
#define HID     768
#define NHEADS  12
#define HDIM    64
#define NBLK    64
#define FFN_    3072
#define NCLS    1500

typedef __attribute__((ext_vector_type(8)))  float          v8f;
typedef __attribute__((ext_vector_type(16))) __bf16         v16bf;
typedef __attribute__((ext_vector_type(8)))  unsigned short u16x8;
typedef __attribute__((ext_vector_type(16))) unsigned short u16x16;

// ---------------- helpers ----------------

__device__ __forceinline__ unsigned short f2bf(float f) {
  unsigned int u = __builtin_bit_cast(unsigned int, f);
  u += 0x7fffu + ((u >> 16) & 1u);            // round-to-nearest-even
  return (unsigned short)(u >> 16);
}

__device__ __forceinline__ v8f vzero8() {
  v8f z;
  #pragma unroll
  for (int i = 0; i < 8; ++i) z[i] = 0.0f;
  return z;
}

__device__ __forceinline__ v8f wmma_bf16(v16bf a, v16bf b, v8f c) {
  return __builtin_amdgcn_wmma_f32_16x16x32_bf16(false, a, false, b, (short)0, c,
                                                 false, false);
}

// low 32 bits of a generic LDS pointer == wave-relative LDS byte address
__device__ __forceinline__ unsigned lds_off(const void* p) {
  return (unsigned)(size_t)p;
}

// CDNA5 async global -> LDS copy, 16 bytes per lane (ASYNCcnt tracked)
__device__ __forceinline__ void async_ld16(unsigned lds_byte, const void* gaddr) {
  asm volatile("global_load_async_to_lds_b128 %0, %1, off"
               :: "v"(lds_byte), "v"(gaddr) : "memory");
}
__device__ __forceinline__ void wait_async0() {
  asm volatile("s_wait_asynccnt 0x0" ::: "memory");
}
__device__ __forceinline__ void wait_async4() {
  asm volatile("s_wait_asynccnt 0x4" ::: "memory");
}

// A fragment (16x32 bf16, M x K) from LDS row-major [M][K].
// lanes 0-15: K 0..7 & 16..23 ; lanes 16-31: K 8..15 & 24..31 (ISA 7.12.2)
__device__ __forceinline__ v16bf afrag_lds(const unsigned short* base, int stride,
                                           int mrow, int kbase, int lane) {
  int row = mrow + (lane & 15);
  int j0  = (lane < 16) ? 0 : 8;
  u16x8 lo = *reinterpret_cast<const u16x8*>(base + row*stride + kbase + j0);
  u16x8 hi = *reinterpret_cast<const u16x8*>(base + row*stride + kbase + 16 + j0);
  u16x16 c = __builtin_shufflevector(lo, hi, 0,1,2,3,4,5,6,7,8,9,10,11,12,13,14,15);
  return __builtin_bit_cast(v16bf, c);
}

// B fragment (32x16 bf16, K x N) from LDS stored transposed [N][K].
// lanes 0-15: K 0..7 & 16..23 ; lanes 16-31: K 8..15 & 24..31 (matches A)
__device__ __forceinline__ v16bf bfrag_lds(const unsigned short* base, int stride,
                                           int nrow, int kbase, int lane) {
  int row = nrow + (lane & 15);
  int k0  = kbase + ((lane < 16) ? 0 : 8);
  u16x8 lo = *reinterpret_cast<const u16x8*>(base + row*stride + k0);
  u16x8 hi = *reinterpret_cast<const u16x8*>(base + row*stride + k0 + 16);
  u16x16 c = __builtin_shufflevector(lo, hi, 0,1,2,3,4,5,6,7,8,9,10,11,12,13,14,15);
  return __builtin_bit_cast(v16bf, c);
}

// A fragment with on-the-fly f32 -> bf16 conversion (for probabilities in LDS)
__device__ __forceinline__ v16bf afrag_f32(const float* base, int stride,
                                           int mrow, int kbase, int lane) {
  int row = mrow + (lane & 15);
  int j0  = (lane < 16) ? 0 : 8;
  const float* p0 = base + row*stride + kbase + j0;
  const float* p1 = base + row*stride + kbase + 16 + j0;
  u16x16 c;
  #pragma unroll
  for (int j = 0; j < 8; ++j) c[j] = f2bf(p0[j]);
  #pragma unroll
  for (int j = 0; j < 8; ++j) c[8+j] = f2bf(p1[j]);
  return __builtin_bit_cast(v16bf, c);
}

__device__ __forceinline__ float gelu_exact(float x) {
  return 0.5f * x * (1.0f + erff(x * 0.70710678118654752f));
}

// ---------------- conversion kernels (run once per launch) ----------------

// plain f32 -> bf16
__global__ __launch_bounds__(256) void cvt_kernel(const float* __restrict__ in,
    unsigned short* __restrict__ out, long n) {
  long i = ((long)blockIdx.x*256 + threadIdx.x)*4;
  if (i >= n) return;
  const float4 f = *reinterpret_cast<const float4*>(in + i);
  out[i+0] = f2bf(f.x); out[i+1] = f2bf(f.y);
  out[i+2] = f2bf(f.z); out[i+3] = f2bf(f.w);
}

// weight f32 [K,N] -> bf16 transposed [N,K]
__global__ __launch_bounds__(256) void wcvt_kernel(const float* __restrict__ W,
    unsigned short* __restrict__ Wt, int K, int N) {
  long idx = (long)blockIdx.x*256 + threadIdx.x;
  if (idx >= (long)K*N) return;
  const int n  = (int)(idx / K);
  const int kk = (int)(idx % K);
  Wt[idx] = f2bf(W[(size_t)kk*N + n]);
}

// -------- GEMM: C[M,N] = act(A[M,K]bf16 @ Bt[N,K]bf16^T + bias) * scale -----
// 256 threads = 8 waves; WG tile 128x128; double-buffered async LDS fills.

template<int ACT, int OUTBF>
__global__ __launch_bounds__(256) void gemm_kernel(
    const unsigned short* __restrict__ A, const unsigned short* __restrict__ B,
    const float* __restrict__ bias, void* __restrict__ Cout,
    int M, int N, int K, float cscale)
{
  __shared__ __align__(16) unsigned short As[2][128*40];
  __shared__ __align__(16) unsigned short Bs[2][128*40];
  const int n0 = blockIdx.x * 128;
  const int m0 = blockIdx.y * 128;
  const int t = threadIdx.x;
  const int lane = t & 31;
  const int w = t >> 5;
  const int wm = (w & 3) * 32;     // wave row base within WG tile
  const int wn = (w >> 2) * 64;    // wave col base within WG tile

  v8f acc[2][4];
  #pragma unroll
  for (int i = 0; i < 2; ++i)
    #pragma unroll
    for (int j = 0; j < 4; ++j) acc[i][j] = vzero8();

  // cooperative async fill: each thread copies 2x16B for A and 2x16B for B
  const int crow = t >> 1;
  const int cp   = (t & 1) * 16;   // element offset of first chunk
  const unsigned short* Arow = A + (size_t)(m0 + crow)*K + cp;
  const unsigned short* Brow = B + (size_t)(n0 + crow)*K + cp;
  const unsigned ldsA0 = lds_off(&As[0][crow*40 + cp]);
  const unsigned ldsB0 = lds_off(&Bs[0][crow*40 + cp]);
  const unsigned bufstep = 128*40*2;   // bytes between double buffers

  // prologue: tile 0
  async_ld16(ldsA0,      Arow);
  async_ld16(ldsA0 + 16, Arow + 8);
  async_ld16(ldsB0,      Brow);
  async_ld16(ldsB0 + 16, Brow + 8);

  int pb = 0;
  for (int k0 = 0; k0 < K; k0 += 32, pb ^= 1) {
    if (k0 + 32 < K) {
      const unsigned lA = ldsA0 + (unsigned)(pb ^ 1)*bufstep;
      const unsigned lB = ldsB0 + (unsigned)(pb ^ 1)*bufstep;
      async_ld16(lA,      Arow + k0 + 32);
      async_ld16(lA + 16, Arow + k0 + 40);
      async_ld16(lB,      Brow + k0 + 32);
      async_ld16(lB + 16, Brow + k0 + 40);
      wait_async4();                 // current tile's 4 ops complete (in-order)
    } else {
      wait_async0();
    }
    __syncthreads();
    const unsigned short* Ab = As[pb];
    const unsigned short* Bb = Bs[pb];
    v16bf a0 = afrag_lds(Ab, 40, wm,      0, lane);
    v16bf a1 = afrag_lds(Ab, 40, wm + 16, 0, lane);
    #pragma unroll
    for (int nt = 0; nt < 4; ++nt) {
      v16bf bf = bfrag_lds(Bb, 40, wn + nt*16, 0, lane);
      acc[0][nt] = wmma_bf16(a0, bf, acc[0][nt]);
      acc[1][nt] = wmma_bf16(a1, bf, acc[1][nt]);
    }
    __syncthreads();
  }

  const int rb = (lane >= 16) ? 8 : 0;
  #pragma unroll
  for (int ms = 0; ms < 2; ++ms) {
    #pragma unroll
    for (int nt = 0; nt < 4; ++nt) {
      const int col = n0 + wn + nt*16 + (lane & 15);
      const float bv = bias[col];
      #pragma unroll
      for (int r = 0; r < 8; ++r) {
        const int row = m0 + wm + ms*16 + rb + r;
        float val = acc[ms][nt][r] + bv;
        if (ACT == 1) val = gelu_exact(val);
        val *= cscale;
        if (OUTBF) ((unsigned short*)Cout)[(size_t)row*N + col] = f2bf(val);
        else       ((float*)Cout)[(size_t)row*N + col] = val;
      }
    }
  }
}

// ---------------- BigBird block-sparse attention (bf16 q/k/v) ----------------

__device__ __forceinline__ int chunk_block(int qi, int c, const int* __restrict__ rnd,
                                           int edge) {
  if (edge) return c;
  if (c < 3)  return qi - 1 + c;      // window
  if (c == 3) return 0;               // global first
  if (c == 4) return NBLK - 1;        // global last
  return rnd[(qi - 1)*3 + (c - 5)];   // random
}

// async-copy a 64x64 bf16 tile (row-major, global ld = HID) into LDS stride 72
__device__ __forceinline__ void tile64_async(unsigned short* lds_tile,
                                             const unsigned short* gbase, int t) {
  const int row = t >> 2;
  const int p0  = (t & 3) * 16;
  const unsigned short* g = gbase + (size_t)row*HID + p0;
  const unsigned l = lds_off(lds_tile + row*72 + p0);
  async_ld16(l,      g);
  async_ld16(l + 16, g + 8);
}

__global__ __launch_bounds__(256) void attn_kernel(
    const unsigned short* __restrict__ q, const unsigned short* __restrict__ k,
    const unsigned short* __restrict__ v, const int* __restrict__ mask,
    const int* __restrict__ rnd, unsigned short* __restrict__ out)
{
  __shared__ __align__(16) unsigned short Qs[64*72];
  __shared__ __align__(16) unsigned short Ks[64*72];
  __shared__ __align__(16) unsigned short Vt[64*72];
  __shared__ __align__(16) float Ss[64*68];
  __shared__ float mterm[64];
  __shared__ float rowm[64];
  __shared__ float rowl[64];

  const int bid = blockIdx.x;
  const int qi = bid % NBLK;
  const int h  = (bid / NBLK) % NHEADS;
  const int b  = bid / (NBLK * NHEADS);
  const int t = threadIdx.x;
  const int lane = t & 31;
  const int w = t >> 5;
  const int mt  = (w & 3) * 16;       // wave's M subtile base
  const int nt0 = (w >> 2) * 32;      // wave's N/D subtile bases: nt0, nt0+16

  const int ldrow = t >> 2;           // cooperative-load row (0..63)
  const int ldcg  = (t & 3) * 16;     // cooperative-load col group

  // Q tile (1/sqrt(d) already folded into the Wq GEMM epilogue)
  tile64_async(Qs, q + (size_t)(b*SEQ + qi*64)*HID + h*HDIM, t);
  if (t < 64) { rowm[t] = -3.0e38f; rowl[t] = 0.0f; }
  wait_async0();
  __syncthreads();

  const v16bf qa0 = afrag_lds(Qs, 72, mt, 0,  lane);
  const v16bf qa1 = afrag_lds(Qs, 72, mt, 32, lane);

  const int edge = (qi == 0 || qi == NBLK-1) ? 1 : 0;
  const int nch = edge ? NBLK : 8;

  // ---- pass 1: global row max & sum-exp (online) ----
  for (int c = 0; c < nch; ++c) {
    const int kb = chunk_block(qi, c, rnd, edge);
    tile64_async(Ks, k + (size_t)(b*SEQ + kb*64)*HID + h*HDIM, t);
    if (t < 64) mterm[t] = ((float)mask[b*SEQ + kb*64 + t] - 1.0f) * 1.0e9f;
    wait_async0();
    __syncthreads();

    v8f s0 = vzero8(), s1 = vzero8();
    #pragma unroll
    for (int ks = 0; ks < 2; ++ks) {
      v16bf aq = ks ? qa1 : qa0;
      v16bf b0 = bfrag_lds(Ks, 72, nt0,      ks*32, lane);
      v16bf b1 = bfrag_lds(Ks, 72, nt0 + 16, ks*32, lane);
      s0 = wmma_bf16(aq, b0, s0);
      s1 = wmma_bf16(aq, b1, s1);
    }
    {
      const int rb = mt + ((lane >= 16) ? 8 : 0);
      const int c0 = nt0 + (lane & 15);
      #pragma unroll
      for (int r = 0; r < 8; ++r) {
        Ss[(rb + r)*68 + c0]      = s0[r];
        Ss[(rb + r)*68 + c0 + 16] = s1[r];
      }
    }
    __syncthreads();

    if (t < 64) {
      float m0 = rowm[t], l0 = rowl[t];
      float cm = -3.0e38f;
      for (int j = 0; j < 64; ++j) cm = fmaxf(cm, Ss[t*68 + j] + mterm[j]);
      float mn = fmaxf(m0, cm);
      float s = 0.0f;
      for (int j = 0; j < 64; ++j) s += __expf(Ss[t*68 + j] + mterm[j] - mn);
      rowl[t] = l0 * __expf(m0 - mn) + s;
      rowm[t] = mn;
    }
    __syncthreads();
  }
  if (t < 64) rowl[t] = (rowl[t] > 0.0f) ? (1.0f / rowl[t]) : 0.0f;
  __syncthreads();

  // ---- pass 2: O += P @ V with final normalization ----
  v8f o0 = vzero8(), o1 = vzero8();
  for (int c = 0; c < nch; ++c) {
    const int kb = chunk_block(qi, c, rnd, edge);
    tile64_async(Ks, k + (size_t)(b*SEQ + kb*64)*HID + h*HDIM, t);
    {
      // V transposed [d][key] (element-granularity -> regular ld + b16 scatter)
      const unsigned short* src = v + (size_t)(b*SEQ + kb*64 + ldrow)*HID + h*HDIM + ldcg;
      u16x8 a0v = *reinterpret_cast<const u16x8*>(src);
      u16x8 a1v = *reinterpret_cast<const u16x8*>(src + 8);
      #pragma unroll
      for (int j = 0; j < 8; ++j) Vt[(ldcg + j)*72 + ldrow] = a0v[j];
      #pragma unroll
      for (int j = 0; j < 8; ++j) Vt[(ldcg + 8 + j)*72 + ldrow] = a1v[j];
    }
    if (t < 64) mterm[t] = ((float)mask[b*SEQ + kb*64 + t] - 1.0f) * 1.0e9f;
    wait_async0();
    __syncthreads();

    v8f s0 = vzero8(), s1 = vzero8();
    #pragma unroll
    for (int ks = 0; ks < 2; ++ks) {
      v16bf aq = ks ? qa1 : qa0;
      v16bf b0 = bfrag_lds(Ks, 72, nt0,      ks*32, lane);
      v16bf b1 = bfrag_lds(Ks, 72, nt0 + 16, ks*32, lane);
      s0 = wmma_bf16(aq, b0, s0);
      s1 = wmma_bf16(aq, b1, s1);
    }
    {
      const int rb = mt + ((lane >= 16) ? 8 : 0);
      const int c0 = nt0 + (lane & 15);
      #pragma unroll
      for (int r = 0; r < 8; ++r) {
        Ss[(rb + r)*68 + c0]      = s0[r];
        Ss[(rb + r)*68 + c0 + 16] = s1[r];
      }
    }
    __syncthreads();

    {
      const float m = rowm[ldrow], il = rowl[ldrow];
      #pragma unroll
      for (int j = 0; j < 16; ++j) {
        const int col = ldcg + j;
        Ss[ldrow*68 + col] = __expf(Ss[ldrow*68 + col] + mterm[col] - m) * il;
      }
    }
    __syncthreads();

    #pragma unroll
    for (int ks = 0; ks < 2; ++ks) {
      v16bf pa = afrag_f32(Ss, 68, mt, ks*32, lane);
      v16bf b0 = bfrag_lds(Vt, 72, nt0,      ks*32, lane);
      v16bf b1 = bfrag_lds(Vt, 72, nt0 + 16, ks*32, lane);
      o0 = wmma_bf16(pa, b0, o0);
      o1 = wmma_bf16(pa, b1, o1);
    }
    __syncthreads();
  }

  {
    unsigned short* dst = out + (size_t)(b*SEQ + qi*64)*HID + h*HDIM;
    const int rb = mt + ((lane >= 16) ? 8 : 0);
    const int c0 = nt0 + (lane & 15);
    #pragma unroll
    for (int r = 0; r < 8; ++r) {
      dst[(size_t)(rb + r)*HID + c0]      = f2bf(o0[r]);
      dst[(size_t)(rb + r)*HID + c0 + 16] = f2bf(o1[r]);
    }
  }
}

// ------- LayerNorm: out(f32) & outb(bf16) = LN(res + add) * g + b -------

__global__ __launch_bounds__(256) void ln_kernel(
    const float* __restrict__ res, const float* __restrict__ add, int addIsPos,
    const float* __restrict__ g, const float* __restrict__ bb,
    float* __restrict__ out, unsigned short* __restrict__ outb)
{
  __shared__ float s1[256], s2[256];
  const int row = blockIdx.x, t = threadIdx.x;
  const float* rp = res + (size_t)row * HID;
  const float* ap = add + (size_t)(addIsPos ? (row % SEQ) : row) * HID;
  const float x0 = rp[t]       + ap[t];
  const float x1 = rp[256 + t] + ap[256 + t];
  const float x2 = rp[512 + t] + ap[512 + t];
  s1[t] = x0 + x1 + x2;
  s2[t] = x0*x0 + x1*x1 + x2*x2;
  __syncthreads();
  for (int o = 128; o > 0; o >>= 1) {
    if (t < o) { s1[t] += s1[t+o]; s2[t] += s2[t+o]; }
    __syncthreads();
  }
  const float mu  = s1[0] * (1.0f/768.0f);
  const float var = s2[0] * (1.0f/768.0f) - mu*mu;
  const float rs  = rsqrtf(var + 1e-12f);
  float* op = out + (size_t)row * HID;
  unsigned short* opb = outb + (size_t)row * HID;
  const float y0 = (x0 - mu) * rs * g[t]       + bb[t];
  const float y1 = (x1 - mu) * rs * g[256 + t] + bb[256 + t];
  const float y2 = (x2 - mu) * rs * g[512 + t] + bb[512 + t];
  op[t] = y0;        op[256 + t] = y1;        op[512 + t] = y2;
  opb[t] = f2bf(y0); opb[256 + t] = f2bf(y1); opb[512 + t] = f2bf(y2);
}

// ---------------- pooling & classifier ----------------

__global__ __launch_bounds__(256) void pool_score_kernel(
    const float* __restrict__ h, const float* __restrict__ pw,
    const float* __restrict__ pb, const int* __restrict__ mask,
    float* __restrict__ scores)
{
  __shared__ float red[256];
  const int row = blockIdx.x, t = threadIdx.x;
  float a = h[(size_t)row*HID + t]       * pw[t]
          + h[(size_t)row*HID + 256 + t] * pw[256 + t]
          + h[(size_t)row*HID + 512 + t] * pw[512 + t];
  red[t] = a; __syncthreads();
  for (int o = 128; o > 0; o >>= 1) { if (t < o) red[t] += red[t+o]; __syncthreads(); }
  if (t == 0) {
    float sc = red[0] + pb[0];
    if (mask[row] == 0) sc = -1.0e9f;
    scores[row] = sc;
  }
}

__global__ __launch_bounds__(256) void pool_softmax_kernel(
    const float* __restrict__ scores, float* __restrict__ wv)
{
  __shared__ float red[256];
  const int b = blockIdx.x, t = threadIdx.x;
  float m = -3.0e38f;
  for (int i = t; i < SEQ; i += 256) m = fmaxf(m, scores[b*SEQ + i]);
  red[t] = m; __syncthreads();
  for (int o = 128; o > 0; o >>= 1) { if (t < o) red[t] = fmaxf(red[t], red[t+o]); __syncthreads(); }
  m = red[0]; __syncthreads();
  float s = 0.0f;
  for (int i = t; i < SEQ; i += 256) s += __expf(scores[b*SEQ + i] - m);
  red[t] = s; __syncthreads();
  for (int o = 128; o > 0; o >>= 1) { if (t < o) red[t] += red[t+o]; __syncthreads(); }
  const float inv = 1.0f / red[0];
  for (int i = t; i < SEQ; i += 256) wv[b*SEQ + i] = __expf(scores[b*SEQ + i] - m) * inv;
}

__global__ __launch_bounds__(256) void pooled_kernel(
    const float* __restrict__ h, const float* __restrict__ wv,
    float* __restrict__ pooled)
{
  const int b = blockIdx.y;
  const int col = blockIdx.x * 256 + threadIdx.x;
  if (col >= HID) return;
  float a = 0.0f;
  for (int s = 0; s < SEQ; ++s)
    a += wv[b*SEQ + s] * h[((size_t)b*SEQ + s)*HID + col];
  pooled[b*HID + col] = a;
}

__global__ __launch_bounds__(256) void cls_kernel(
    const float* __restrict__ pooled, const float* __restrict__ W,
    const float* __restrict__ bias, float* __restrict__ out)
{
  const int idx = blockIdx.x * 256 + threadIdx.x;
  if (idx >= BATCH * NCLS) return;
  const int b = idx / NCLS, n = idx % NCLS;
  float a = bias[n];
  for (int j = 0; j < HID; ++j) a += pooled[b*HID + j] * W[(size_t)j*NCLS + n];
  out[idx] = a;
}

// ---------------- numpy RandomState(1234) rand_attn replication ----------------

__global__ void rand_kernel(int* __restrict__ rnd) {
  if (threadIdx.x != 0 || blockIdx.x != 0) return;
  unsigned int mt[624];
  int mti;
  mt[0] = 1234u;
  for (int i = 1; i < 624; ++i)
    mt[i] = 1812433253u * (mt[i-1] ^ (mt[i-1] >> 30)) + (unsigned int)i;
  mti = 624;
  auto next = [&]() -> unsigned int {
    if (mti >= 624) {
      for (int i = 0; i < 624; ++i) {
        unsigned int y = (mt[i] & 0x80000000u) | (mt[(i+1) % 624] & 0x7fffffffu);
        mt[i] = mt[(i+397) % 624] ^ (y >> 1) ^ ((y & 1u) ? 2567483615u : 0u);
      }
      mti = 0;
    }
    unsigned int y = mt[mti++];
    y ^= y >> 11;
    y ^= (y << 7)  & 2636928640u;
    y ^= (y << 15) & 4022730752u;
    y ^= y >> 18;
    return y;
  };
  auto interval = [&](unsigned int mx) -> unsigned int {  // uniform [0, mx]
    if (mx == 0) return 0;
    unsigned int msk = mx;
    msk |= msk >> 1; msk |= msk >> 2; msk |= msk >> 4; msk |= msk >> 8; msk |= msk >> 16;
    unsigned int vv;
    while ((vv = (next() & msk)) > mx) {}
    return vv;
  };
  for (int i = 1; i <= NBLK - 2; ++i) {
    int choices[NBLK]; int np = 0;
    for (int bch = 0; bch < NBLK; ++bch) {
      if (bch == 0 || bch == NBLK-1 || bch == i-1 || bch == i || bch == i+1) continue;
      choices[np++] = bch;
    }
    int perm[NBLK];
    for (int j = 0; j < np; ++j) perm[j] = j;
    for (int j = np - 1; j >= 1; --j) {          // numpy rk_shuffle
      unsigned int r = interval((unsigned int)j);
      int tmpv = perm[j]; perm[j] = perm[r]; perm[r] = tmpv;
    }
    for (int r = 0; r < 3; ++r) rnd[(i-1)*3 + r] = choices[perm[r]];
  }
}

// ---------------- host orchestration ----------------

extern "C" void kernel_launch(void* const* d_in, const int* in_sizes, int n_in,
                              void* d_out, int out_size, void* d_ws, size_t ws_size,
                              hipStream_t stream)
{
  (void)in_sizes; (void)n_in; (void)out_size; (void)ws_size;
  const float* x     = (const float*)d_in[0];
  const int*   mask  = (const int*)  d_in[1];
  const float* projW = (const float*)d_in[2];
  const float* projB = (const float*)d_in[3];
  const float* embG  = (const float*)d_in[4];
  const float* embB  = (const float*)d_in[5];
  const float* posE  = (const float*)d_in[6];
  const float* poolW = (const float*)d_in[7];
  const float* poolB = (const float*)d_in[8];
  const float* clsW  = (const float*)d_in[9];
  const float* clsB  = (const float*)d_in[10];

  char* ws = (char*)d_ws;
  size_t off = 0;
  auto walloc = [&](size_t bytes) -> void* {
    void* p = ws + off;
    off = (off + bytes + 255) & ~(size_t)255;
    return p;
  };
  // f32 masters
  float* h  = (float*)walloc((size_t)TOK*HID*4);
  float* t1 = (float*)walloc((size_t)TOK*HID*4);
  float* t2 = (float*)walloc((size_t)TOK*HID*4);
  // bf16 activations
  unsigned short* xb  = (unsigned short*)walloc((size_t)TOK*1024*2);
  unsigned short* hb  = (unsigned short*)walloc((size_t)TOK*HID*2);
  unsigned short* qb  = (unsigned short*)walloc((size_t)TOK*HID*2);
  unsigned short* kbq = (unsigned short*)walloc((size_t)TOK*HID*2);
  unsigned short* vbq = (unsigned short*)walloc((size_t)TOK*HID*2);
  unsigned short* ab  = (unsigned short*)walloc((size_t)TOK*HID*2);
  unsigned short* f1b = (unsigned short*)walloc((size_t)TOK*FFN_*2);
  // bf16 transposed weights (re-converted per layer; trivial traffic)
  unsigned short* projWt = (unsigned short*)walloc((size_t)1024*HID*2);
  unsigned short* wqT = (unsigned short*)walloc((size_t)HID*HID*2);
  unsigned short* wkT = (unsigned short*)walloc((size_t)HID*HID*2);
  unsigned short* wvT = (unsigned short*)walloc((size_t)HID*HID*2);
  unsigned short* woT = (unsigned short*)walloc((size_t)HID*HID*2);
  unsigned short* w1T = (unsigned short*)walloc((size_t)HID*FFN_*2);
  unsigned short* w2T = (unsigned short*)walloc((size_t)FFN_*HID*2);
  // pooling / misc
  float* scores = (float*)walloc((size_t)TOK*4);
  float* wv     = (float*)walloc((size_t)TOK*4);
  float* pooled = (float*)walloc((size_t)BATCH*HID*4);
  int*   rnd    = (int*)  walloc((size_t)(NBLK-2)*3*4);

  rand_kernel<<<1, 1, 0, stream>>>(rnd);

  auto nblk = [](long n) -> int { return (int)((n + 255) / 256); };

  cvt_kernel<<<nblk((long)TOK*1024/4), 256, 0, stream>>>(x, xb, (long)TOK*1024);
  wcvt_kernel<<<nblk((long)1024*HID), 256, 0, stream>>>(projW, projWt, 1024, HID);

  const dim3 g768(HID/128, TOK/128);    // (6, 128)
  const dim3 g3072(FFN_/128, TOK/128);  // (24, 128)

  // embed: h = LN(x @ projW + projB + pos_emb)
  gemm_kernel<0,0><<<g768, 256, 0, stream>>>(xb, projWt, projB, (void*)t1,
                                             TOK, HID, 1024, 1.0f);
  ln_kernel<<<TOK, 256, 0, stream>>>(t1, posE, 1, embG, embB, h, hb);

  for (int L = 0; L < 12; ++L) {
    const int base = 11 + L*16;
    const float* Wq = (const float*)d_in[base+0];
    const float* bq = (const float*)d_in[base+1];
    const float* Wk = (const float*)d_in[base+2];
    const float* bk_= (const float*)d_in[base+3];
    const float* Wv = (const float*)d_in[base+4];
    const float* bv_= (const float*)d_in[base+5];
    const float* Wo = (const float*)d_in[base+6];
    const float* bo = (const float*)d_in[base+7];
    const float* W1 = (const float*)d_in[base+8];
    const float* b1 = (const float*)d_in[base+9];
    const float* W2 = (const float*)d_in[base+10];
    const float* b2 = (const float*)d_in[base+11];
    const float* g1 = (const float*)d_in[base+12];
    const float* be1= (const float*)d_in[base+13];
    const float* g2 = (const float*)d_in[base+14];
    const float* be2= (const float*)d_in[base+15];

    wcvt_kernel<<<nblk((long)HID*HID), 256, 0, stream>>>(Wq, wqT, HID, HID);
    wcvt_kernel<<<nblk((long)HID*HID), 256, 0, stream>>>(Wk, wkT, HID, HID);
    wcvt_kernel<<<nblk((long)HID*HID), 256, 0, stream>>>(Wv, wvT, HID, HID);
    wcvt_kernel<<<nblk((long)HID*HID), 256, 0, stream>>>(Wo, woT, HID, HID);
    wcvt_kernel<<<nblk((long)HID*FFN_), 256, 0, stream>>>(W1, w1T, HID, FFN_);
    wcvt_kernel<<<nblk((long)FFN_*HID), 256, 0, stream>>>(W2, w2T, FFN_, HID);

    // q scaled by 1/sqrt(HDIM) in the epilogue
    gemm_kernel<0,1><<<g768, 256, 0, stream>>>(hb, wqT, bq,  (void*)qb,
                                               TOK, HID, HID, 0.125f);
    gemm_kernel<0,1><<<g768, 256, 0, stream>>>(hb, wkT, bk_, (void*)kbq,
                                               TOK, HID, HID, 1.0f);
    gemm_kernel<0,1><<<g768, 256, 0, stream>>>(hb, wvT, bv_, (void*)vbq,
                                               TOK, HID, HID, 1.0f);
    attn_kernel<<<BATCH*NHEADS*NBLK, 256, 0, stream>>>(qb, kbq, vbq, mask, rnd, ab);
    gemm_kernel<0,0><<<g768, 256, 0, stream>>>(ab, woT, bo, (void*)t2,
                                               TOK, HID, HID, 1.0f);
    ln_kernel<<<TOK, 256, 0, stream>>>(h, t2, 0, g1, be1, h, hb);
    gemm_kernel<1,1><<<g3072, 256, 0, stream>>>(hb, w1T, b1, (void*)f1b,
                                                TOK, FFN_, HID, 1.0f);
    gemm_kernel<0,0><<<g768, 256, 0, stream>>>(f1b, w2T, b2, (void*)t2,
                                               TOK, HID, FFN_, 1.0f);
    ln_kernel<<<TOK, 256, 0, stream>>>(h, t2, 0, g2, be2, h, hb);
  }

  pool_score_kernel<<<TOK, 256, 0, stream>>>(h, poolW, poolB, mask, scores);
  pool_softmax_kernel<<<BATCH, 256, 0, stream>>>(scores, wv);
  pooled_kernel<<<dim3(HID/256, BATCH), 256, 0, stream>>>(h, wv, pooled);
  cls_kernel<<<(BATCH*NCLS + 255)/256, 256, 0, stream>>>(pooled, clsW, clsB,
                                                         (float*)d_out);
}
